// QLV4_Linear_MOD_38946763440196
// MI455X (gfx1250) — matrix operand order repaired and checked by reading, not compiled, not run on hardware
//
#include <hip/hip_runtime.h>

typedef __attribute__((ext_vector_type(16))) __bf16 v16bf;
typedef __attribute__((ext_vector_type(8)))  __bf16 v8bf;
typedef __attribute__((ext_vector_type(8)))  float  v8f;
typedef __attribute__((__vector_size__(16))) int    i32x4;

#define AS1 __attribute__((address_space(1)))
#define AS3 __attribute__((address_space(3)))

#define IN_F   8192
#define OUT_F  8192
#define GROUPS_PER_ROW 64     // 8192 / 128
#define M_TOTAL 1024

#define BM 128
#define BN 128
#define BK 128                // one quant group per K-tile
#define KTILES (IN_F / BK)    // 64
#define LDA (BK + 8)          // ushort units, pad for LDS banks
#define LDB (BK + 8)

#if defined(__gfx1250__) && __has_builtin(__builtin_amdgcn_global_load_async_to_lds_b128)
#define HAVE_ASYNC_LDS 1
#else
#define HAVE_ASYNC_LDS 0
#endif

__device__ __forceinline__ void wait_asynccnt_8() {
#if __has_builtin(__builtin_amdgcn_s_wait_asynccnt)
  __builtin_amdgcn_s_wait_asynccnt(8);
#else
  asm volatile("s_wait_asynccnt 8" ::: "memory");
#endif
}
__device__ __forceinline__ void wait_asynccnt_0() {
#if __has_builtin(__builtin_amdgcn_s_wait_asynccnt)
  __builtin_amdgcn_s_wait_asynccnt(0);
#else
  asm volatile("s_wait_asynccnt 0" ::: "memory");
#endif
}

// Fence-free workgroup barrier: no vmcnt/loadcnt drain, so register-prefetched
// globals stay in flight across it. Memory clobber = compiler-level fence only.
__device__ __forceinline__ void wg_barrier() {
  asm volatile("s_barrier_signal -1\n\ts_barrier_wait -1" ::: "memory");
}
// Producer-side LDS visibility before the rendezvous.
__device__ __forceinline__ void wait_lds_stores() {
  asm volatile("s_wait_dscnt 0" ::: "memory");
}

// A fragment (16-bit 16x32): lane holds M=lane%16; two 16B chunks at k, k+16
__device__ __forceinline__ v16bf load_a_frag(const unsigned short* p) {
  v8bf lo = *(const v8bf*)p;
  v8bf hi = *(const v8bf*)(p + 16);
  v16bf r;
#pragma unroll
  for (int i = 0; i < 8; ++i) { r[i] = lo[i]; r[i + 8] = hi[i]; }
  return r;
}
// B fragment (32x16 KxN): lane holds N=lane%16; two 16B chunks at k, k+16
__device__ __forceinline__ v16bf load_b_frag(const unsigned short* p) {
  v8bf lo = *(const v8bf*)p;
  v8bf hi = *(const v8bf*)(p + 16);
  v16bf r;
#pragma unroll
  for (int i = 0; i < 8; ++i) { r[i] = lo[i]; r[i + 8] = hi[i]; }
  return r;
}

__global__ __launch_bounds__(256) void qlinear_wmma_bf16(
    const unsigned short* __restrict__ x,   // bf16 bits [1024][8192]
    const float* __restrict__ wq,           // int4 codes as f32 [8192][8192]
    const float* __restrict__ wscale,       // [524288]
    const float* __restrict__ wzp,          // [524288]
    float* __restrict__ out)                // f32 [1024][8192]
{
  __shared__ unsigned short lsA[2][BM * LDA];  // double-buffered (async target)
  __shared__ unsigned short lsB[BN * LDB];

  const int tid = threadIdx.x;
  const int m0 = blockIdx.x * BM;   // M-tile fastest -> L2 reuse of weight tiles
  const int n0 = blockIdx.y * BN;

  const int lane = tid & 31;
  const int r16  = lane & 15;
  const int hi   = lane >> 4;
  const int wv   = tid >> 5;
  const int wm   = (wv & 3) * 32;
  const int wn   = (wv >> 2) * 64;

  // cooperative loader coords: 2 threads per row, 64 elems each
  const int lrow = tid >> 1;
  const int lcol = (tid & 1) << 6;

  const unsigned short* xrow = x  + (size_t)(m0 + lrow) * IN_F + lcol;
  const float*          wrow = wq + (size_t)(n0 + lrow) * IN_F + lcol;
  const float*          srow = wscale + (size_t)(n0 + lrow) * GROUPS_PER_ROW;
  const float*          zrow = wzp    + (size_t)(n0 + lrow) * GROUPS_PER_ROW;

  float4 breg[16];          // staged fp32 weight codes for current K-tile
  float  s_st = 0.f, z_st = 0.f;
#if !HAVE_ASYNC_LDS
  uint4  areg[8];
#endif

  auto prefetchB = [&](int kt) {
    const float4* bsrc = (const float4*)(wrow + kt * BK);
#pragma unroll
    for (int i = 0; i < 16; ++i) breg[i] = bsrc[i];
    s_st = srow[kt];
    z_st = zrow[kt];
  };
#if HAVE_ASYNC_LDS
  auto issueA = [&](int kt, int buf) {   // 8 x async 16B global->LDS per thread
    const unsigned short* asrc = xrow + kt * BK;
    unsigned loff = (unsigned)(unsigned long long)&lsA[buf][lrow * LDA + lcol];
#pragma unroll
    for (int i = 0; i < 8; ++i)
      __builtin_amdgcn_global_load_async_to_lds_b128(
          (AS1 i32x4*)(unsigned long long)(asrc + i * 8),
          (AS3 i32x4*)(unsigned long long)(loff + i * 16),
          0, 0);
  };
#else
  auto loadA = [&](int kt) {
    const uint4* asrc = (const uint4*)(xrow + kt * BK);
#pragma unroll
    for (int i = 0; i < 8; ++i) areg[i] = asrc[i];
  };
#endif

  v8f acc[2][4] = {};

  // ---- prologue: tile 0 in flight ----
  prefetchB(0);
#if HAVE_ASYNC_LDS
  issueA(0, 0);
#else
  loadA(0);
#endif

  for (int kt = 0; kt < KTILES; ++kt) {
    const int buf = kt & 1;

    // kick next tile's async A first: flies during the dequant VALU burst
#if HAVE_ASYNC_LDS
    if (kt + 1 < KTILES) issueA(kt + 1, buf ^ 1);  // buffer fenced by trailing barrier
#endif

    // dequant + store staged B codes: bf16( fma(q, s, -z*s) )
    // (__bf16)fmaf(..) into packed halves -> v_fma_mixlo/mixhi_bf16 (1 VALU/elem)
    {
      const float s  = s_st;
      const float ns = -z_st * s_st;
      unsigned short* dst = &lsB[lrow * LDB + lcol];
#pragma unroll
      for (int i = 0; i < 8; ++i) {
        float4 q0 = breg[2 * i];
        float4 q1 = breg[2 * i + 1];
        v8bf pk;
        pk[0] = (__bf16)fmaf(q0.x, s, ns);
        pk[1] = (__bf16)fmaf(q0.y, s, ns);
        pk[2] = (__bf16)fmaf(q0.z, s, ns);
        pk[3] = (__bf16)fmaf(q0.w, s, ns);
        pk[4] = (__bf16)fmaf(q1.x, s, ns);
        pk[5] = (__bf16)fmaf(q1.y, s, ns);
        pk[6] = (__bf16)fmaf(q1.z, s, ns);
        pk[7] = (__bf16)fmaf(q1.w, s, ns);
        *(v8bf*)(dst + i * 8) = pk;
      }
    }
#if !HAVE_ASYNC_LDS
    {
      uint4* dst = (uint4*)&lsA[buf][lrow * LDA + lcol];
#pragma unroll
      for (int i = 0; i < 8; ++i) dst[i] = areg[i];
    }
#endif

    // next tile's B codes into registers; stay in flight through compute
    if (kt + 1 < KTILES) {
#if !HAVE_ASYNC_LDS
      loadA(kt + 1);
#endif
      prefetchB(kt + 1);
    }

#if HAVE_ASYNC_LDS
    if (kt + 1 < KTILES) wait_asynccnt_8();  // current tile's 8 done (in-order)
    else                 wait_asynccnt_0();
#endif
    wait_lds_stores();
    wg_barrier();

    // ---- compute: 4 WMMA k-steps of 32 ----
#pragma unroll
    for (int ks = 0; ks < 4; ++ks) {
      const int kb = ks * 32;
      v16bf a0 = load_a_frag(&lsA[buf][(wm      + r16) * LDA + kb + hi * 8]);
      v16bf a1 = load_a_frag(&lsA[buf][(wm + 16 + r16) * LDA + kb + hi * 8]);
#pragma unroll
      for (int ni = 0; ni < 4; ++ni) {
        v16bf b = load_b_frag(&lsB[(wn + ni * 16 + r16) * LDB + kb + hi * 8]);
        acc[0][ni] = __builtin_amdgcn_wmma_f32_16x16x32_bf16(
            false, a0, false, b, (short)0, acc[0][ni], false, false);
        acc[1][ni] = __builtin_amdgcn_wmma_f32_16x16x32_bf16(
            false, a1, false, b, (short)0, acc[1][ni], false, false);
      }
    }

    wg_barrier();  // fences lsB overwrite + next async A issue
  }

  // ---- store (C/D layout: N = lane%16, M = (lane/16)*8 + r), streaming ----
#pragma unroll
  for (int mi = 0; mi < 2; ++mi) {
#pragma unroll
    for (int r = 0; r < 8; ++r) {
      const int m = m0 + wm + mi * 16 + hi * 8 + r;
      float* orow = out + (size_t)m * OUT_F + n0 + wn + r16;
#pragma unroll
      for (int ni = 0; ni < 4; ++ni)
        __builtin_nontemporal_store(acc[mi][ni][r], orow + ni * 16);
    }
  }
}

extern "C" void kernel_launch(void* const* d_in, const int* in_sizes, int n_in,
                              void* d_out, int out_size, void* d_ws, size_t ws_size,
                              hipStream_t stream) {
  const unsigned short* x  = (const unsigned short*)d_in[0];  // bf16 input
  const float* wq          = (const float*)d_in[1];
  const float* wscale      = (const float*)d_in[2];
  const float* wzp         = (const float*)d_in[3];
  float* out               = (float*)d_out;

  dim3 grid(M_TOTAL / BM, OUT_F / BN);   // x = M-tile (fast) for L2 weight reuse
  qlinear_wmma_bf16<<<grid, 256, 0, stream>>>(x, wq, wscale, wzp, out);
}
